// RNNHead_17128329576685
// MI455X (gfx1250) — matrix operand order repaired and checked by reading, not compiled, not run on hardware
//
#include <hip/hip_runtime.h>
#include <hip/hip_bf16.h>

typedef __attribute__((ext_vector_type(16))) _Float16 v16h;
typedef __attribute__((ext_vector_type(8)))  float    v8f;
typedef __attribute__((ext_vector_type(4)))  float    fvec4;

#define LOG2E 1.44269504088896340736f

__device__ __forceinline__ float fast_sigmoid(float v) {
    float e = __builtin_amdgcn_exp2f(-LOG2E * v);
    return __builtin_amdgcn_rcpf(1.0f + e);
}
__device__ __forceinline__ float fast_tanh(float v) {
    float e = __builtin_amdgcn_exp2f(-2.0f * LOG2E * v);
    return (1.0f - e) * __builtin_amdgcn_rcpf(1.0f + e);
}

// One wave32 per 16-row batch tile. Weights register-resident in WMMA B-layout.
// Biases pre-broadcast into loop-invariant WMMA C accumulators (no bias adds in
// the loop); r/z gates chain the h-projection WMMA onto the x-projection
// accumulator (no merge adds). h carried as f32 D-layout regs + f16 A-layout
// copy rebuilt each step via a padded 16x36 LDS transpose tile.
__global__ __launch_bounds__(32) void gru_head_fused(
    const float* __restrict__ x, const float* __restrict__ W_ih,
    const float* __restrict__ W_hh, const float* __restrict__ b_ih,
    const float* __restrict__ b_hh, const float* __restrict__ W_head,
    const float* __restrict__ b_head, float* __restrict__ out)
{
    constexpr int T = 256, F = 64, H = 32;
    const int lane = threadIdx.x;          // 0..31
    const int lo = lane & 15;              // N / row index within tile
    const int hi = lane >> 4;              // half-wave select
    const int b0 = blockIdx.x * 16;        // batch tile base

    __shared__ float hbuf[16][36];         // padded: stride 36 floats (144B)

    // ---- preload W_ih^T as B-tiles: element e <-> K = kt*32 + (e<8 ? 8*hi+e : 16+8*hi+e-8), N = nt*16+lo
    v16h wih[12];
    #pragma unroll
    for (int nt = 0; nt < 6; ++nt) {
        #pragma unroll
        for (int kt = 0; kt < 2; ++kt) {
            const float* p = W_ih + (nt * 16 + lo) * F + kt * 32 + hi * 8;
            v16h b;
            #pragma unroll
            for (int e = 0; e < 16; ++e) b[e] = (_Float16)p[e < 8 ? e : e + 8];
            wih[nt * 2 + kt] = b;
        }
    }
    // ---- preload W_hh^T as B-tiles: K = (e<8 ? 8*hi+e : 16+8*hi+e-8), N = nt*16+lo
    v16h whh[6];
    #pragma unroll
    for (int nt = 0; nt < 6; ++nt) {
        const float* p = W_hh + (nt * 16 + lo) * H + hi * 8;
        v16h b;
        #pragma unroll
        for (int e = 0; e < 16; ++e) b[e] = (_Float16)p[e < 8 ? e : e + 8];
        whh[nt] = b;
    }

    // ---- biases broadcast into loop-invariant C accumulators (column nt*16+lo)
    v8f c_rz[4];   // r gate (nt=0,1), z gate (nt=2,3): b_ih + b_hh combined
    v8f c_nx[2];   // n gate x-part (nt=4,5): b_ih only
    v8f c_nh[2];   // n gate h-part (nt=4,5): b_hh only
    #pragma unroll
    for (int nt = 0; nt < 4; ++nt) {
        float s = b_ih[nt * 16 + lo] + b_hh[nt * 16 + lo];
        #pragma unroll
        for (int e = 0; e < 8; ++e) c_rz[nt][e] = s;
    }
    #pragma unroll
    for (int q = 0; q < 2; ++q) {
        float sx = b_ih[(4 + q) * 16 + lo];
        float sh = b_hh[(4 + q) * 16 + lo];
        #pragma unroll
        for (int e = 0; e < 8; ++e) { c_nx[q][e] = sx; c_nh[q][e] = sh; }
    }

    v16h hA = {};          // h in f16 A-layout (row l%16, K per A-map)
    float hprev[8][2];     // h in f32 D-layout: row v+8*hi, col p*16+lo
    #pragma unroll
    for (int v = 0; v < 8; ++v) { hprev[v][0] = 0.0f; hprev[v][1] = 0.0f; }

    const float* xrow = x + (size_t)(b0 + lo) * T * F;

    #pragma unroll 2
    for (int t = 0; t < T; ++t) {
        // ---- load x tile directly into f16 A-layout (two K-tiles of 32)
        const float* xb = xrow + (size_t)t * F;
        v16h xA[2];
        #pragma unroll
        for (int kt = 0; kt < 2; ++kt) {
            fvec4 a0 = *(const fvec4*)(xb + kt * 32 + hi * 8);
            fvec4 a1 = *(const fvec4*)(xb + kt * 32 + hi * 8 + 4);
            fvec4 a2 = *(const fvec4*)(xb + kt * 32 + 16 + hi * 8);
            fvec4 a3 = *(const fvec4*)(xb + kt * 32 + 16 + hi * 8 + 4);
            v16h va;
            #pragma unroll
            for (int e = 0; e < 4; ++e) {
                va[e]      = (_Float16)a0[e];
                va[e + 4]  = (_Float16)a1[e];
                va[e + 8]  = (_Float16)a2[e];
                va[e + 12] = (_Float16)a3[e];
            }
            xA[kt] = va;
        }

        // ---- r/z gates: bias-init C, x-proj (2 WMMAs) then h-proj chained (1 WMMA)
        v8f acc_rz[4];
        #pragma unroll
        for (int nt = 0; nt < 4; ++nt) {
            v8f a = c_rz[nt];
            a = __builtin_amdgcn_wmma_f32_16x16x32_f16(
                    false, xA[0], false, wih[2 * nt],     (short)0, a, false, false);
            a = __builtin_amdgcn_wmma_f32_16x16x32_f16(
                    false, xA[1], false, wih[2 * nt + 1], (short)0, a, false, false);
            a = __builtin_amdgcn_wmma_f32_16x16x32_f16(
                    false, hA,    false, whh[nt],         (short)0, a, false, false);
            acc_rz[nt] = a;
        }
        // ---- n gate: x-part and h-part kept separate (h-part is scaled by r)
        v8f acc_nx[2], acc_nh[2];
        #pragma unroll
        for (int q = 0; q < 2; ++q) {
            v8f ax = c_nx[q];
            ax = __builtin_amdgcn_wmma_f32_16x16x32_f16(
                    false, xA[0], false, wih[2 * (4 + q)],     (short)0, ax, false, false);
            ax = __builtin_amdgcn_wmma_f32_16x16x32_f16(
                    false, xA[1], false, wih[2 * (4 + q) + 1], (short)0, ax, false, false);
            acc_nx[q] = ax;
            acc_nh[q] = __builtin_amdgcn_wmma_f32_16x16x32_f16(
                    false, hA, false, whh[4 + q], (short)0, c_nh[q], false, false);
        }

        // ---- gate math + store new h to LDS in row-major [16][36]
        #pragma unroll
        for (int v = 0; v < 8; ++v) {
            #pragma unroll
            for (int p = 0; p < 2; ++p) {
                float r  = fast_sigmoid(acc_rz[p][v]);
                float zg = fast_sigmoid(acc_rz[2 + p][v]);
                float n  = fast_tanh(fmaf(r, acc_nh[p][v], acc_nx[p][v]));
                float hn = fmaf(zg, hprev[v][p] - n, n);   // (1-z)*n + z*h
                hprev[v][p] = hn;
                hbuf[v + 8 * hi][p * 16 + lo] = hn;
            }
        }
        __syncthreads();   // single-wave WG: lowers to S_NOP; forces DS ordering

        // ---- transpose-reload h into f16 A-layout for the next step
        {
            const float* row = &hbuf[lo][0];
            fvec4 a0 = *(const fvec4*)(row + hi * 8);
            fvec4 a1 = *(const fvec4*)(row + hi * 8 + 4);
            fvec4 a2 = *(const fvec4*)(row + 16 + hi * 8);
            fvec4 a3 = *(const fvec4*)(row + 16 + hi * 8 + 4);
            v16h va;
            #pragma unroll
            for (int e = 0; e < 4; ++e) {
                va[e]      = (_Float16)a0[e];
                va[e + 4]  = (_Float16)a1[e];
                va[e + 8]  = (_Float16)a2[e];
                va[e + 12] = (_Float16)a3[e];
            }
            hA = va;
        }
        __syncthreads();
    }

    // ---- head: out[b] = h_last . W_head + b_head
    float w0 = W_head[lo];
    float w1 = W_head[16 + lo];
    float bb = b_head[0];
    #pragma unroll
    for (int v = 0; v < 8; ++v) {
        float part = hprev[v][0] * w0 + hprev[v][1] * w1;
        #pragma unroll
        for (int m = 1; m < 16; m <<= 1)
            part += __shfl_xor(part, m, 32);
        if (lo == 0) out[b0 + 8 * hi + v] = part + bb;
    }
}

extern "C" void kernel_launch(void* const* d_in, const int* in_sizes, int n_in,
                              void* d_out, int out_size, void* d_ws, size_t ws_size,
                              hipStream_t stream) {
    const float* x      = (const float*)d_in[0];   // [4096, 256, 64]
    const float* W_ih   = (const float*)d_in[1];   // [96, 64]
    const float* W_hh   = (const float*)d_in[2];   // [96, 32]
    const float* b_ih   = (const float*)d_in[3];   // [96]
    const float* b_hh   = (const float*)d_in[4];   // [96]
    const float* W_head = (const float*)d_in[5];   // [1, 32]
    const float* b_head = (const float*)d_in[6];   // [1]
    float* out = (float*)d_out;                    // [4096, 1]

    constexpr int B = 4096;
    dim3 grid(B / 16);
    dim3 block(32);
    gru_head_fused<<<grid, block, 0, stream>>>(x, W_ih, W_hh, b_ih, b_hh,
                                               W_head, b_head, out);
}
